// DecoderRNN_74354473828828
// MI455X (gfx1250) — compile-verified, matrix operand order checked by reading, not run on hardware
//
#include <hip/hip_runtime.h>
#include <hip/hip_bf16.h>
#include <stddef.h>

// ---------------------------------------------------------------------------
// DecoderRNN (LSTM caption decoder) for MI455X / gfx1250, bf16 WMMA pipeline.
// B=32, T=64, E=H=512, V=32000.
// ---------------------------------------------------------------------------

typedef __bf16 bf16_t;
typedef __attribute__((ext_vector_type(16))) __bf16 v16bf;
typedef __attribute__((ext_vector_type(8)))  float  v8f;

#define B_        32
#define T_        64
#define EH_       512          // E == H == 512
#define G4_       2048         // 4*H
#define V_        32000
#define ROWS_     2016         // (T-1)*B
#define ROWS_PAD_ 2048
#define NWG_LSTM_ 32

// ---- workspace layout (bytes) ----
static const size_t OFF_WIH  = 0;                                   // 2048*512 bf16
static const size_t OFF_WHH  = OFF_WIH  + (size_t)G4_ * EH_ * 2;    // 2048*512 bf16
static const size_t OFF_FCW  = OFF_WHH  + (size_t)G4_ * EH_ * 2;    // 32000*512 bf16
static const size_t OFF_X    = OFF_FCW  + (size_t)V_  * EH_ * 2;    // 2048*512 bf16 (padded)
static const size_t OFF_HS   = OFF_X    + (size_t)ROWS_PAD_ * EH_ * 2; // 2048*512 bf16 (padded)
static const size_t OFF_HBUF = OFF_HS   + (size_t)ROWS_PAD_ * EH_ * 2; // 2*32*512 bf16
static const size_t OFF_GPRE = OFF_HBUF + (size_t)2 * B_ * EH_ * 2; // 2048*2048 f32
static const size_t OFF_BARS = OFF_GPRE + (size_t)ROWS_PAD_ * G4_ * 4; // 64 u32

// ---------------------------------------------------------------------------
// WMMA helpers
// ---------------------------------------------------------------------------
__device__ __forceinline__ v16bf load_frag(const bf16_t* __restrict__ base,
                                           int ld, int row0, int k0, int lane) {
  // lane 0..15 -> K block [k0, k0+16); lane 16..31 -> [k0+16, k0+32)
  const bf16_t* p = base + (size_t)(row0 + (lane & 15)) * ld + k0 + ((lane & 16) ? 16 : 0);
  return *(const v16bf*)p;
}

__device__ __forceinline__ v8f wmma_bf16(v16bf a, v16bf b, v8f c) {
  return __builtin_amdgcn_wmma_f32_16x16x32_bf16(false, a, false, b, (short)0, c,
                                                 false, false);
}

__device__ __forceinline__ float sigmoidf_(float x) {
  return 1.0f / (1.0f + __expf(-x));
}

// ---------------------------------------------------------------------------
// 1) f32 -> bf16 conversion (grid-stride)
// ---------------------------------------------------------------------------
__global__ void f32_to_bf16_kernel(const float* __restrict__ in,
                                   bf16_t* __restrict__ out, int n) {
  int i = blockIdx.x * blockDim.x + threadIdx.x;
  int stride = gridDim.x * blockDim.x;
  for (; i < n; i += stride) out[i] = (bf16_t)in[i];
}

// ---------------------------------------------------------------------------
// 2) Build x-sequence [2048(pad) x 512] bf16:
//    row r = t*32 + b ; t==0 -> features[b], t>=1 -> embed[captions[b][t+1]]
// ---------------------------------------------------------------------------
__global__ void build_x_kernel(const float* __restrict__ features,
                               const int*   __restrict__ captions,
                               const float* __restrict__ embed,
                               bf16_t* __restrict__ X) {
  int gid = blockIdx.x * blockDim.x + threadIdx.x;   // 2048*512 = 1,048,576 threads
  int r = gid >> 9;
  int e = gid & (EH_ - 1);
  if (r >= ROWS_PAD_) return;
  float v = 0.0f;
  if (r < ROWS_) {
    int t = r >> 5;          // 0..62
    int b = r & 31;
    if (t == 0) {
      v = features[(size_t)b * EH_ + e];
    } else {
      int tok = captions[b * T_ + (t + 1)];
      v = embed[(size_t)tok * EH_ + e];
    }
  }
  X[(size_t)r * EH_ + e] = (bf16_t)v;
}

// ---------------------------------------------------------------------------
// 3) Init: zero out[:,0,:], h0 buffer, Hs pad rows, barrier counters
// ---------------------------------------------------------------------------
__global__ void init_kernel(float* __restrict__ out,
                            bf16_t* __restrict__ hbuf0,
                            bf16_t* __restrict__ hs_pad,
                            unsigned* __restrict__ bars) {
  int tid = blockIdx.x * blockDim.x + threadIdx.x;
  int stride = gridDim.x * blockDim.x;
  for (int i = tid; i < B_ * V_; i += stride) {
    int b = i / V_; int v = i % V_;
    out[(size_t)b * T_ * V_ + v] = 0.0f;          // out[:,0,:] = 0
  }
  for (int i = tid; i < B_ * EH_; i += stride) hbuf0[i] = (bf16_t)0.0f;
  for (int i = tid; i < (ROWS_PAD_ - ROWS_) * EH_; i += stride) hs_pad[i] = (bf16_t)0.0f;
  for (int i = tid; i < 64; i += stride) bars[i] = 0u;
}

// ---------------------------------------------------------------------------
// 4) gates_pre = X @ W_ih^T + (b_ih + b_hh)    -> [2048 x 2048] f32
//    WG tile 64x256, 8 waves, wave tile = 2 Mt x 4 Nt, K=512
// ---------------------------------------------------------------------------
__global__ __launch_bounds__(256) void gates_pre_kernel(
    const bf16_t* __restrict__ X, const bf16_t* __restrict__ Wih,
    const float* __restrict__ bih, const float* __restrict__ bhh,
    float* __restrict__ gpre) {
  int lane = threadIdx.x & 31;
  int wid  = threadIdx.x >> 5;
  int r0   = blockIdx.y * 64;
  int n0wg = blockIdx.x * 256;
  int mts  = (wid & 1) * 2;
  int nts  = (wid >> 1) * 4;

  v8f acc[2][4];
#pragma unroll
  for (int j = 0; j < 4; ++j) {
    int n = n0wg + (nts + j) * 16 + (lane & 15);
    float bv = bih[n] + bhh[n];
#pragma unroll
    for (int i = 0; i < 2; ++i)
#pragma unroll
      for (int v = 0; v < 8; ++v) acc[i][j][v] = bv;
  }

  for (int k0 = 0; k0 < EH_; k0 += 32) {
    v16bf a[2];
#pragma unroll
    for (int i = 0; i < 2; ++i)
      a[i] = load_frag(X, EH_, r0 + (mts + i) * 16, k0, lane);
#pragma unroll
    for (int j = 0; j < 4; ++j) {
      v16bf b = load_frag(Wih, EH_, n0wg + (nts + j) * 16, k0, lane);
#pragma unroll
      for (int i = 0; i < 2; ++i) acc[i][j] = wmma_bf16(a[i], b, acc[i][j]);
    }
  }

#pragma unroll
  for (int i = 0; i < 2; ++i)
#pragma unroll
    for (int j = 0; j < 4; ++j)
#pragma unroll
      for (int v = 0; v < 8; ++v) {
        int r = r0 + (mts + i) * 16 + v + ((lane & 16) ? 8 : 0);
        int c = n0wg + (nts + j) * 16 + (lane & 15);
        gpre[(size_t)r * G4_ + c] = acc[i][j][v];
      }
}

// ---------------------------------------------------------------------------
// 5) Cooperative LSTM recurrence: 32 WGs x 256 threads, 63 steps.
//    WG owns 16 hidden units (cols j0..j0+15 of each of the 4 gates).
//    Wave tile: mt = wid&1 (batch 16-rows), g = wid>>1 (gate). h broadcast via
//    double-buffered global bf16 buffer + atomic/spin grid barrier per step.
// ---------------------------------------------------------------------------
__global__ __launch_bounds__(256) void lstm_kernel(
    const float* __restrict__ gpre, const bf16_t* __restrict__ Whh,
    bf16_t* __restrict__ hbuf, bf16_t* __restrict__ Hs,
    unsigned* __restrict__ bars) {
  __shared__ float gl[4][32][16];   // [gate][batch][hidden-within-slice]

  int lane = threadIdx.x & 31;
  int wid  = threadIdx.x >> 5;
  int j0   = blockIdx.x * 16;       // hidden slice
  int mt   = wid & 1;
  int g    = wid >> 1;              // 0..3
  int m0   = mt * 16;
  int n0   = g * EH_ + j0;          // gate column base in [0,2048)

  float cst[2] = {0.0f, 0.0f};      // cell state: 2 (b,j) pairs per thread

  for (int t = 0; t < T_ - 1; ++t) {
    const bf16_t* hprev = hbuf + (size_t)(t & 1) * (B_ * EH_);

    v8f acc;
#pragma unroll
    for (int v = 0; v < 8; ++v)
      acc[v] = gpre[(size_t)(t * B_ + m0 + v + ((lane & 16) ? 8 : 0)) * G4_ +
                    n0 + (lane & 15)];

    for (int k0 = 0; k0 < EH_; k0 += 32) {
      v16bf a = load_frag(hprev, EH_, m0, k0, lane);
      v16bf b = load_frag(Whh,   EH_, n0, k0, lane);
      acc = wmma_bf16(a, b, acc);
    }

#pragma unroll
    for (int v = 0; v < 8; ++v)
      gl[g][m0 + v + ((lane & 16) ? 8 : 0)][lane & 15] = acc[v];
    __syncthreads();

    bf16_t* hout = hbuf + (size_t)((t + 1) & 1) * (B_ * EH_);
#pragma unroll
    for (int u = 0; u < 2; ++u) {
      int s  = threadIdx.x + u * 256;   // 0..511
      int b  = s >> 4;
      int jj = s & 15;
      float iv = gl[0][b][jj];
      float fv = gl[1][b][jj];
      float gv = gl[2][b][jj];
      float ov = gl[3][b][jj];
      float cn = sigmoidf_(fv) * cst[u] + sigmoidf_(iv) * tanhf(gv);
      cst[u] = cn;
      float h = sigmoidf_(ov) * tanhf(cn);
      bf16_t hb = (bf16_t)h;
      hout[b * EH_ + j0 + jj] = hb;
      Hs[(size_t)(t * B_ + b) * EH_ + j0 + jj] = hb;
    }

    // ---- device-wide barrier for step t ----
    __threadfence();
    __syncthreads();
    if (threadIdx.x == 0) {
      atomicAdd(&bars[t], 1u);
      volatile unsigned* vc = &bars[t];
      while (*vc < (unsigned)NWG_LSTM_) __builtin_amdgcn_s_sleep(1);
    }
    __syncthreads();
    __threadfence();
  }
}

// ---------------------------------------------------------------------------
// 6) logits = Hs @ fc_w^T + fc_b, scattered into out[b][t+1][v].
//    WG tile 64x256 over [2016 x 32000], 8 waves, wave = 2 Mt x 4 Nt.
// ---------------------------------------------------------------------------
__global__ __launch_bounds__(256) void logits_kernel(
    const bf16_t* __restrict__ Hs, const bf16_t* __restrict__ fcw,
    const float* __restrict__ fcb, float* __restrict__ out) {
  int lane = threadIdx.x & 31;
  int wid  = threadIdx.x >> 5;
  int r0   = blockIdx.y * 64;       // 0..1984
  int n0wg = blockIdx.x * 256;      // 0..31744
  int mts  = (wid & 1) * 2;
  int nts  = (wid >> 1) * 4;

  v8f acc[2][4];
#pragma unroll
  for (int j = 0; j < 4; ++j) {
    float bv = fcb[n0wg + (nts + j) * 16 + (lane & 15)];
#pragma unroll
    for (int i = 0; i < 2; ++i)
#pragma unroll
      for (int v = 0; v < 8; ++v) acc[i][j][v] = bv;
  }

  for (int k0 = 0; k0 < EH_; k0 += 32) {
    if (k0 + 32 < EH_) {
      // prefetch next K-slab of B operand (emits global_prefetch_b8)
      const bf16_t* pf = fcw + (size_t)(n0wg + nts * 16 + (lane & 15)) * EH_ +
                         (k0 + 32) + ((lane & 16) ? 16 : 0);
      __builtin_prefetch((const void*)pf, 0, 1);
    }
    v16bf a[2];
#pragma unroll
    for (int i = 0; i < 2; ++i)
      a[i] = load_frag(Hs, EH_, r0 + (mts + i) * 16, k0, lane);
#pragma unroll
    for (int j = 0; j < 4; ++j) {
      v16bf b = load_frag(fcw, EH_, n0wg + (nts + j) * 16, k0, lane);
#pragma unroll
      for (int i = 0; i < 2; ++i) acc[i][j] = wmma_bf16(a[i], b, acc[i][j]);
    }
  }

#pragma unroll
  for (int i = 0; i < 2; ++i)
#pragma unroll
    for (int j = 0; j < 4; ++j)
#pragma unroll
      for (int v = 0; v < 8; ++v) {
        int r = r0 + (mts + i) * 16 + v + ((lane & 16) ? 8 : 0);
        if (r < ROWS_) {
          int t = r >> 5;
          int b = r & 31;
          int col = n0wg + (nts + j) * 16 + (lane & 15);
          out[(size_t)b * T_ * V_ + (size_t)(t + 1) * V_ + col] = acc[i][j][v];
        }
      }
}

// ---------------------------------------------------------------------------
// launcher
// ---------------------------------------------------------------------------
extern "C" void kernel_launch(void* const* d_in, const int* in_sizes, int n_in,
                              void* d_out, int out_size, void* d_ws, size_t ws_size,
                              hipStream_t stream) {
  const float* features = (const float*)d_in[0];
  const int*   captions = (const int*)d_in[1];
  const float* embed    = (const float*)d_in[2];
  const float* Wih      = (const float*)d_in[3];
  const float* Whh      = (const float*)d_in[4];
  const float* bih      = (const float*)d_in[5];
  const float* bhh      = (const float*)d_in[6];
  const float* fcw      = (const float*)d_in[7];
  const float* fcb      = (const float*)d_in[8];
  float* out = (float*)d_out;

  char* ws = (char*)d_ws;
  bf16_t*   wih_b  = (bf16_t*)(ws + OFF_WIH);
  bf16_t*   whh_b  = (bf16_t*)(ws + OFF_WHH);
  bf16_t*   fcw_b  = (bf16_t*)(ws + OFF_FCW);
  bf16_t*   x_b    = (bf16_t*)(ws + OFF_X);
  bf16_t*   hs_b   = (bf16_t*)(ws + OFF_HS);
  bf16_t*   hbuf   = (bf16_t*)(ws + OFF_HBUF);
  float*    gpre   = (float*)(ws + OFF_GPRE);
  unsigned* bars   = (unsigned*)(ws + OFF_BARS);

  // 1) weight conversions to bf16
  f32_to_bf16_kernel<<<1024, 256, 0, stream>>>(Wih, wih_b, G4_ * EH_);
  f32_to_bf16_kernel<<<1024, 256, 0, stream>>>(Whh, whh_b, G4_ * EH_);
  f32_to_bf16_kernel<<<4096, 256, 0, stream>>>(fcw, fcw_b, V_ * EH_);

  // 2) x-sequence (embedding gather + features) in bf16
  build_x_kernel<<<(ROWS_PAD_ * EH_) / 256, 256, 0, stream>>>(features, captions,
                                                              embed, x_b);

  // 3) init: out[:,0,:]=0, h0=0, Hs pad=0, barriers=0
  init_kernel<<<1024, 256, 0, stream>>>(out, hbuf, hs_b + (size_t)ROWS_ * EH_, bars);

  // 4) gates_pre = X @ W_ih^T + b_ih + b_hh
  gates_pre_kernel<<<dim3(G4_ / 256, ROWS_PAD_ / 64), 256, 0, stream>>>(
      x_b, wih_b, bih, bhh, gpre);

  // 5) cooperative LSTM recurrence (32 WGs, per-step grid barrier)
  lstm_kernel<<<NWG_LSTM_, 256, 0, stream>>>(gpre, whh_b, hbuf, hs_b, bars);

  // 6) vocab projection + scatter into [B, T, V]
  logits_kernel<<<dim3(V_ / 256, ROWS_PAD_ / 64), 256, 0, stream>>>(hs_b, fcw_b,
                                                                    fcb, out);
}
